// EinMask_43301860279008
// MI455X (gfx1250) — matrix operand order repaired and checked by reading, not compiled, not run on hardware
//
#include <hip/hip_runtime.h>
#include <math.h>

// ---------------------------------------------------------------- types
typedef __attribute__((ext_vector_type(8)))  float  v8f;
typedef __attribute__((ext_vector_type(16))) __bf16 v16bf;

#define D_    768
#define NH_   12
#define HD_   64
#define N_    512
#define L_    8
#define BE_   16
#define BB_   8
#define EE_   2
#define DQ_   2304
#define MLP_  3072
#define MROWS 8192            // BE_*N_

__device__ __forceinline__ unsigned short f2bf_us(float f) {
  unsigned u = __float_as_uint(f);
  return (unsigned short)((u + 0x7FFFu + ((u >> 16) & 1u)) >> 16);
}
__device__ __forceinline__ __bf16 us2bf(unsigned short s) {
  union { unsigned short u; __bf16 b; } v; v.u = s; return v.b;
}
__device__ __forceinline__ __bf16 f2bf(float f) { return us2bf(f2bf_us(f)); }

// CDNA5 async global->LDS copy: 16 bytes per lane, per-lane LDS dest offset.
__device__ __forceinline__ void async_copy_b128(unsigned ldsOff, const void* g) {
  unsigned long long ga = (unsigned long long)(size_t)g;
  asm volatile("global_load_async_to_lds_b128 %0, %1, off"
               :: "v"(ldsOff), "v"(ga) : "memory");
}
__device__ __forceinline__ void wait_async0() {
  asm volatile("s_wait_asynccnt 0x0" ::: "memory");
}
__device__ __forceinline__ unsigned lds_off(const void* p) {
  return (unsigned)(size_t)p;
}

// fast tanh-GELU via exp
__device__ __forceinline__ float gelu_f(float x) {
  float u  = 0.7978845608f * (x + 0.044715f * x * x * x);
  float e  = __expf(2.f * u);
  float th = 1.f - 2.f / (e + 1.f);
  return 0.5f * x * (1.f + th);
}

// ---------------------------------------------------------------- f32 -> bf16 (row-major)
__global__ void mvit_cvt_kernel(const float* __restrict__ s,
                                unsigned short* __restrict__ d, long n) {
  long i = (long)blockIdx.x * blockDim.x + threadIdx.x;
  if (i < n) d[i] = f2bf_us(s[i]);
}

// ---------------------------------------------------------------- weight pack:
// f32 (K x N) row-major  ->  bf16 fragment-tile order
__global__ void mvit_packB_kernel(const float* __restrict__ src,
                                  unsigned short* __restrict__ dst,
                                  int K, int N) {
  long i = (long)blockIdx.x * blockDim.x + threadIdx.x;
  if (i >= (long)K * N) return;
  int  e    = (int)(i & 15);
  int  lane = (int)((i >> 4) & 31);
  long tile = i >> 9;
  int  KT   = K >> 5;
  int  ct   = (int)(tile / KT);
  int  kt   = (int)(tile % KT);
  int  nl = lane & 15, hf = lane >> 4;
  int  k  = (kt << 5) + (hf << 3) + (e & 7) + ((e >> 3) << 4);
  int  n  = (ct << 4) + nl;
  dst[i] = f2bf_us(src[(long)k * N + n]);
}

// ---------------------------------------------------------------- embed
__global__ void mvit_embed_kernel(const float* __restrict__ fields,
                                  const int*   __restrict__ visible,
                                  const float* __restrict__ W_in,
                                  const float* __restrict__ noise_b,
                                  const float* __restrict__ mask_emb,
                                  const float* __restrict__ pos_emb,
                                  float* __restrict__ X) {
  long idx = (long)blockIdx.x * blockDim.x + threadIdx.x;
  if (idx >= (long)MROWS * D_) return;
  int d  = (int)(idx % D_);
  long t = idx / D_;
  int n  = (int)(t % N_);
  int be = (int)(t / N_);
  int bb = be / EE_;
  int gh = n >> 5, gw = n & 31;
  float v;
  if (visible[bb * N_ + n]) {
    float acc = 0.f;
    for (int vv = 0; vv < 2; ++vv)
      for (int p = 0; p < 4; ++p)
        for (int q = 0; q < 4; ++q)
          acc += fields[((bb * 2 + vv) * 64 + gh * 4 + p) * 128 + gw * 4 + q] *
                 W_in[((vv * 4 + p) * 4 + q) * D_ + d];
    v = acc;
  } else {
    v = mask_emb[d];
  }
  X[idx] = v + noise_b[d] + pos_emb[n * D_ + d];
}

// ---------------------------------------------------------------- layernorm -> bf16
__global__ __launch_bounds__(256) void mvit_ln_kernel(const float* __restrict__ X,
                                                      const float* __restrict__ s,
                                                      const float* __restrict__ b,
                                                      unsigned short* __restrict__ H) {
  __shared__ float sh[512];
  int row = blockIdx.x;
  const float* x = X + (long)row * D_;
  int t = threadIdx.x;
  float ls = 0.f, lq = 0.f;
  for (int i = t; i < D_; i += 256) { float v = x[i]; ls += v; lq += v * v; }
  sh[t] = ls; sh[256 + t] = lq; __syncthreads();
  for (int o = 128; o > 0; o >>= 1) {
    if (t < o) { sh[t] += sh[t + o]; sh[256 + t] += sh[256 + t + o]; }
    __syncthreads();
  }
  float mu  = sh[0] * (1.f / D_);
  float var = sh[256] * (1.f / D_) - mu * mu;
  float inv = rsqrtf(var + 1e-5f);
  unsigned short* h = H + (long)row * D_;
  for (int i = t; i < D_; i += 256)
    h[i] = f2bf_us((x[i] - mu) * inv * s[i] + b[i]);
}

// ---------------------------------------------------------------- WMMA GEMM
#define BM_ 128
#define BN_ 64

__global__ __launch_bounds__(256) void mvit_gemm_bf16(
    const unsigned short* __restrict__ A, const unsigned short* __restrict__ Bp,
    const float* __restrict__ bias, const float* res,
    float* outF, unsigned short* outB,
    int M, int N, int K, int gelu) {
  __shared__ __align__(32) unsigned short As[2][8 * 512];
  __shared__ __align__(32) unsigned short Bs[2][4 * 512];

  const int t = threadIdx.x;
  const int wave = t >> 5, lane = t & 31;
  const int half = lane >> 4, ln = lane & 15;
  const int wm = wave >> 1, wn = wave & 1;
  const int blockM = blockIdx.y * BM_, blockN = blockIdx.x * BN_;
  const int KT = K >> 5;

  const int arow = t >> 1, aseg = (t & 1) * 16;
  const int am = arow & 15, atm = arow >> 4;
  const unsigned ldsA0 = (unsigned)((atm * 512 + am * 16 + (aseg ? 8 : 0)) * 2);
  const int bct = t >> 6, binner = (t & 63) * 8;
  const unsigned ldsB0 = (unsigned)((bct * 512 + binner) * 2);
  const long bTileBase = ((long)((blockN >> 4) + bct) * KT) * 512 + binner;

  const unsigned baseA0 = lds_off(&As[0][0]), baseA1 = lds_off(&As[1][0]);
  const unsigned baseB0 = lds_off(&Bs[0][0]), baseB1 = lds_off(&Bs[1][0]);

  v8f z8 = {0.f,0.f,0.f,0.f,0.f,0.f,0.f,0.f};
  v8f acc[2][2];
  acc[0][0] = z8; acc[0][1] = z8; acc[1][0] = z8; acc[1][1] = z8;

  auto stage = [&](int kt, int buf) {
    const unsigned bA = buf ? baseA1 : baseA0;
    const unsigned bB = buf ? baseB1 : baseB0;
    const unsigned short* ga = A + (long)(blockM + arow) * K + (kt << 5) + aseg;
    async_copy_b128(bA + ldsA0,       ga);
    async_copy_b128(bA + ldsA0 + 512, ga + 8);
    async_copy_b128(bB + ldsB0, Bp + bTileBase + (long)kt * 512);
  };

  stage(0, 0);
  for (int kt = 0; kt < KT; ++kt) {
    const int buf = kt & 1;
    wait_async0();
    __syncthreads();
    if (kt + 1 < KT) stage(kt + 1, buf ^ 1);

    const unsigned short* Ab = &As[buf][0];
    const unsigned short* Bb = &Bs[buf][0];
    v16bf af[2], bfr[2];
    #pragma unroll
    for (int i = 0; i < 2; ++i)
      af[i] = *reinterpret_cast<const v16bf*>(Ab + (wm * 2 + i) * 512 + lane * 16);
    #pragma unroll
    for (int j = 0; j < 2; ++j)
      bfr[j] = *reinterpret_cast<const v16bf*>(Bb + (wn * 2 + j) * 512 + lane * 16);
    #pragma unroll
    for (int i = 0; i < 2; ++i)
      #pragma unroll
      for (int jj = 0; jj < 2; ++jj)
        acc[i][jj] = __builtin_amdgcn_wmma_f32_16x16x32_bf16(
            false, af[i], false, bfr[jj], (short)0, acc[i][jj], false, false);
    __syncthreads();
  }

  #pragma unroll
  for (int i = 0; i < 2; ++i)
    #pragma unroll
    for (int jj = 0; jj < 2; ++jj) {
      int col = blockN + wn * 32 + jj * 16 + ln;
      float bv = bias ? bias[col] : 0.f;
      #pragma unroll
      for (int r = 0; r < 8; ++r) {
        int row = blockM + wm * 32 + i * 16 + r + 8 * half;
        float v = acc[i][jj][r] + bv;
        if (gelu) v = gelu_f(v);
        long o = (long)row * N + col;
        if (res)  v += res[o];
        if (outF) outF[o] = v;
        if (outB) outB[o] = f2bf_us(v);
      }
    }
}

// ---------------------------------------------------------------- fused attention
// Q staged in A-fragment order, K staged in B-fragment order (async copies);
// V staged row-major (async); P fragments converted f32->bf16 from softmax LDS.
#define AR_ 16
__global__ __launch_bounds__(256) void mvit_attn_kernel(
    const unsigned short* __restrict__ QKV, unsigned short* __restrict__ O) {
  __shared__ __align__(32) unsigned short Qf[2 * 512];     // A-frag tiles (K=64)
  __shared__ __align__(32) unsigned short KVf[8192];       // K frags / V rows
  __shared__ float Sm[AR_ * 512];
  __shared__ float red[256];

  const int t = threadIdx.x;
  const int rb = blockIdx.x, h = blockIdx.y, be = blockIdx.z;
  const int wave = t >> 5, lane = t & 31, half = lane >> 4, ln = lane & 15;
  const int wquad = wave & 3;
  const int wsel  = wave >> 2;

  const long base = ((long)be * N_) * DQ_ + (long)h * HD_;
  const unsigned qBase  = lds_off(&Qf[0]);
  const unsigned kvBase = lds_off(&KVf[0]);

  // K staging indices: thread -> one K row, 32 d-values (two 16-value chunks)
  const int kn  = t >> 1;               // row within 128-block
  const int ksg = (t & 1) * 32;         // d offset 0/32
  const int kct = kn >> 4, knl = kn & 15;
  auto stageK = [&](int jbo) {
    const unsigned short* src = QKV + base + (long)(jbo * 128 + kn) * DQ_ + 768 + ksg;
    #pragma unroll
    for (int c = 0; c < 2; ++c) {
      int c16 = (ksg >> 4) + c;
      int kc = c16 >> 1, hf = c16 & 1;
      unsigned dst = kvBase + (unsigned)((((kct * 2 + kc) * 32 + knl + 16 * hf) * 16) * 2);
      async_copy_b128(dst,      src + c * 16);
      async_copy_b128(dst + 16, src + c * 16 + 8);
    }
  };
  // V staging: row-major [128][64]
  auto stageV = [&](int kbo) {
    const unsigned short* src = QKV + base + (long)(kbo * 128 + kn) * DQ_ + 1536 + ksg;
    unsigned dst = kvBase + (unsigned)((kn * 64 + ksg) * 2);
    #pragma unroll
    for (int c = 0; c < 4; ++c)
      async_copy_b128(dst + c * 16, src + c * 8);
  };

  // Q staging (threads 0..127): one 8-value group each, lane-swizzled dest
  if (t < 128) {
    int r = t >> 3, g = t & 7;
    int kt = g >> 2, gg = g & 3;
    unsigned dst = qBase +
        (unsigned)(((kt * 512 + (r + 16 * (gg >> 1)) * 16 + (gg & 1) * 8)) * 2);
    async_copy_b128(dst, QKV + base + (long)(rb * AR_ + r) * DQ_ + g * 8);
  }
  stageK(0);

  // ---- S = Q K^T * scale ----
  for (int jbo = 0; jbo < 4; ++jbo) {
    wait_async0();
    __syncthreads();
    v8f sacc = {0.f,0.f,0.f,0.f,0.f,0.f,0.f,0.f};
    #pragma unroll
    for (int kc = 0; kc < 2; ++kc) {
      v16bf af = *reinterpret_cast<const v16bf*>(&Qf[kc * 512 + lane * 16]);
      int tile = (wsel * 4 + wquad) * 2 + kc;
      v16bf bf = *reinterpret_cast<const v16bf*>(&KVf[tile * 512 + lane * 16]);
      sacc = __builtin_amdgcn_wmma_f32_16x16x32_bf16(false, af, false, bf,
                                                     (short)0, sacc, false, false);
    }
    int col = (jbo * 2 + wsel) * 64 + wquad * 16 + ln;
    #pragma unroll
    for (int r = 0; r < 8; ++r) Sm[(r + 8 * half) * 512 + col] = sacc[r] * 0.125f;
    __syncthreads();
    if (jbo < 3) stageK(jbo + 1);
  }

  // overlap: start streaming V block 0 while softmax runs
  stageV(0);

  // ---- softmax ----
  {
    int row = t >> 4, sub = t & 15;
    float* srow = &Sm[row * 512];
    float mx = -1e30f;
    for (int c = sub; c < 512; c += 16) mx = fmaxf(mx, srow[c]);
    red[t] = mx; __syncthreads();
    for (int o = 8; o > 0; o >>= 1) { if (sub < o) red[t] = fmaxf(red[t], red[t + o]); __syncthreads(); }
    mx = red[row * 16];
    __syncthreads();
    float s = 0.f;
    for (int c = sub; c < 512; c += 16) { float e = __expf(srow[c] - mx); srow[c] = e; s += e; }
    red[t] = s; __syncthreads();
    for (int o = 8; o > 0; o >>= 1) { if (sub < o) red[t] += red[t + o]; __syncthreads(); }
    float inv = 1.f / red[row * 16];
    for (int c = sub; c < 512; c += 16) srow[c] *= inv;
  }

  // ---- O = P @ V (split-K across wave pairs) ----
  v8f oacc = {0.f,0.f,0.f,0.f,0.f,0.f,0.f,0.f};
  for (int kbo = 0; kbo < 4; ++kbo) {
    wait_async0();
    __syncthreads();
    int kb = kbo * 2 + wsel;
    int cl = wquad * 16 + ln;
    #pragma unroll
    for (int kc = 0; kc < 2; ++kc) {
      v16bf af, bf;
      const float* prow = &Sm[ln * 512 + kb * 64 + kc * 32];
      #pragma unroll
      for (int j = 0; j < 4; ++j) {
        int ka = half * 16 + 2 * j, kb2 = half * 16 + 8 + 2 * j;
        af[2*j]   = f2bf(prow[ka]);  af[2*j+1] = f2bf(prow[ka + 1]);
        af[8+2*j] = f2bf(prow[kb2]); af[9+2*j] = f2bf(prow[kb2 + 1]);
      }
      #pragma unroll
      for (int j = 0; j < 8; ++j) {
        int kk = wsel * 64 + kc * 32 + half * 16 + 2 * j;
        bf[2*j]   = us2bf(KVf[kk * 64 + cl]);
        bf[2*j+1] = us2bf(KVf[(kk + 1) * 64 + cl]);
      }
      oacc = __builtin_amdgcn_wmma_f32_16x16x32_bf16(false, af, false, bf,
                                                     (short)0, oacc, false, false);
    }
    __syncthreads();
    if (kbo < 3) stageV(kbo + 1);
  }

  __syncthreads();
  int cl = wquad * 16 + ln;
  if (wsel == 1) {
    #pragma unroll
    for (int r = 0; r < 8; ++r) Sm[(r + 8 * half) * 64 + cl] = oacc[r];
  }
  __syncthreads();
  if (wsel == 0) {
    #pragma unroll
    for (int r = 0; r < 8; ++r) {
      float v = oacc[r] + Sm[(r + 8 * half) * 64 + cl];
      int n = rb * AR_ + r + 8 * half;
      O[((long)be * N_ + n) * D_ + h * HD_ + cl] = f2bf_us(v);
    }
  }
}

// ---------------------------------------------------------------- un-patch
__global__ void mvit_unpatch_kernel(const float* __restrict__ X,
                                    const float* __restrict__ W_out,
                                    float* __restrict__ out) {
  long o = (long)blockIdx.x * blockDim.x + threadIdx.x;
  if (o >= (long)BB_ * 2 * 64 * 128 * EE_) return;
  int e  = (int)(o % EE_); long t = o / EE_;
  int ww = (int)(t % 128); t /= 128;
  int hh = (int)(t % 64);  t /= 64;
  int v  = (int)(t % 2);
  int bb = (int)(t / 2);
  int be = bb * EE_ + e;
  int gh = hh >> 2, p = hh & 3, gw = ww >> 2, q = ww & 3;
  int n  = gh * 32 + gw;
  const float* xr = X + ((long)be * N_ + n) * D_;
  const float* w  = W_out + (long)v * D_ * 16 + p * 4 + q;
  float acc = 0.f;
  for (int d = 0; d < D_; ++d) acc += xr[d] * w[d * 16];
  out[o] = acc;
}

// ---------------------------------------------------------------- host
extern "C" void kernel_launch(void* const* d_in, const int* in_sizes, int n_in,
                              void* d_out, int out_size, void* d_ws, size_t ws_size,
                              hipStream_t stream) {
  const float* fields   = (const float*)d_in[0];
  const int*   visible  = (const int*)  d_in[1];
  const float* z        = (const float*)d_in[2];
  const float* W_in     = (const float*)d_in[3];
  const float* W_out    = (const float*)d_in[4];
  const float* noise_W  = (const float*)d_in[5];
  const float* noise_b  = (const float*)d_in[6];
  const float* mask_emb = (const float*)d_in[7];
  const float* pos_emb  = (const float*)d_in[8];
  const float* ln1_s    = (const float*)d_in[9];
  const float* ln1_b    = (const float*)d_in[10];
  const float* Wqkv     = (const float*)d_in[11];
  const float* bqkv     = (const float*)d_in[12];
  const float* Wo       = (const float*)d_in[13];
  const float* bo       = (const float*)d_in[14];
  const float* ln2_s    = (const float*)d_in[15];
  const float* ln2_b    = (const float*)d_in[16];
  const float* W1       = (const float*)d_in[17];
  const float* b1       = (const float*)d_in[18];
  const float* W2       = (const float*)d_in[19];
  const float* b2       = (const float*)d_in[20];
  (void)in_sizes; (void)n_in; (void)out_size; (void)ws_size;

  size_t off = 0;
  auto take = [&](size_t elems, size_t esz) -> char* {
    char* p = (char*)d_ws + off;
    off += (elems * esz + 255) & ~(size_t)255;
    return p;
  };
  unsigned short* wn_bf   = (unsigned short*)take((size_t)D_ * D_, 2);
  unsigned short* wqkv_bf = (unsigned short*)take((size_t)L_ * D_ * DQ_, 2);
  unsigned short* wo_bf   = (unsigned short*)take((size_t)L_ * D_ * D_, 2);
  unsigned short* w1_bf   = (unsigned short*)take((size_t)L_ * D_ * MLP_, 2);
  unsigned short* w2_bf   = (unsigned short*)take((size_t)L_ * D_ * MLP_, 2);
  unsigned short* zbf     = (unsigned short*)take((size_t)MROWS * D_, 2);
  float*          Xf      = (float*)         take((size_t)MROWS * D_, 4);
  unsigned short* Hbf     = (unsigned short*)take((size_t)MROWS * D_, 2);
  unsigned short* QKVbf   = (unsigned short*)take((size_t)MROWS * DQ_, 2);
  unsigned short* AObf    = (unsigned short*)take((size_t)MROWS * D_, 2);
  unsigned short* Hmlp    = (unsigned short*)take((size_t)MROWS * MLP_, 2);

  auto packB = [&](const float* s, unsigned short* d, int K, int N) {
    long n = (long)K * N;
    mvit_packB_kernel<<<(unsigned)((n + 255) / 256), 256, 0, stream>>>(s, d, K, N);
  };
  packB(noise_W, wn_bf, D_, D_);
  for (int l = 0; l < L_; ++l) {
    packB(Wqkv + (size_t)l * D_ * DQ_,  wqkv_bf + (size_t)l * D_ * DQ_,  D_,   DQ_);
    packB(Wo   + (size_t)l * D_ * D_,   wo_bf   + (size_t)l * D_ * D_,   D_,   D_);
    packB(W1   + (size_t)l * D_ * MLP_, w1_bf   + (size_t)l * D_ * MLP_, D_,   MLP_);
    packB(W2   + (size_t)l * D_ * MLP_, w2_bf   + (size_t)l * D_ * MLP_, MLP_, D_);
  }
  mvit_cvt_kernel<<<(unsigned)(((long)MROWS * D_ + 255) / 256), 256, 0, stream>>>(
      z, zbf, (long)MROWS * D_);

  mvit_embed_kernel<<<(unsigned)(((long)MROWS * D_ + 255) / 256), 256, 0, stream>>>(
      fields, visible, W_in, noise_b, mask_emb, pos_emb, Xf);

  // X += z @ noise_W
  mvit_gemm_bf16<<<dim3(D_ / BN_, MROWS / BM_), 256, 0, stream>>>(
      zbf, wn_bf, nullptr, Xf, Xf, nullptr, MROWS, D_, D_, 0);

  for (int l = 0; l < L_; ++l) {
    mvit_ln_kernel<<<MROWS, 256, 0, stream>>>(Xf, ln1_s + l * D_, ln1_b + l * D_, Hbf);

    mvit_gemm_bf16<<<dim3(DQ_ / BN_, MROWS / BM_), 256, 0, stream>>>(
        Hbf, wqkv_bf + (size_t)l * D_ * DQ_, bqkv + l * DQ_,
        nullptr, nullptr, QKVbf, MROWS, DQ_, D_, 0);

    mvit_attn_kernel<<<dim3(N_ / AR_, NH_, BE_), 256, 0, stream>>>(QKVbf, AObf);

    mvit_gemm_bf16<<<dim3(D_ / BN_, MROWS / BM_), 256, 0, stream>>>(
        AObf, wo_bf + (size_t)l * D_ * D_, bo + l * D_,
        Xf, Xf, nullptr, MROWS, D_, D_, 0);

    mvit_ln_kernel<<<MROWS, 256, 0, stream>>>(Xf, ln2_s + l * D_, ln2_b + l * D_, Hbf);

    mvit_gemm_bf16<<<dim3(MLP_ / BN_, MROWS / BM_), 256, 0, stream>>>(
        Hbf, w1_bf + (size_t)l * D_ * MLP_, b1 + l * MLP_,
        nullptr, nullptr, Hmlp, MROWS, MLP_, D_, 1);

    mvit_gemm_bf16<<<dim3(D_ / BN_, MROWS / BM_), 256, 0, stream>>>(
        Hmlp, w2_bf + (size_t)l * D_ * MLP_, b2 + l * D_,
        Xf, Xf, nullptr, MROWS, D_, MLP_, 0);
  }

  mvit_unpatch_kernel<<<(unsigned)(((long)BB_ * 2 * 64 * 128 * EE_ + 255) / 256),
                        256, 0, stream>>>(Xf, W_out, (float*)d_out);
}
